// QMamba_84293028151651
// MI455X (gfx1250) — matrix operand order repaired and checked by reading, not compiled, not run on hardware
//
#include <hip/hip_runtime.h>

// ---------------- model constants (match reference) ----------------
constexpr int kDIM = 768;    // d_model
constexpr int kDIN = 1024;   // vision feature dim
constexpr int kNQ  = 384;    // queries
constexpr int kNB  = 12;     // blocks
constexpr int kB   = 2;      // batch
constexpr int kL   = 1024;   // image tokens
constexpr int kDI  = 1536;   // mamba d_inner
constexpr int kDS  = 16;     // d_state
constexpr int kDC  = 4;      // d_conv
constexpr int kDTR = 48;     // dt_rank
constexpr int kNH  = 16;     // heads
constexpr int kHD  = 48;     // head dim
constexpr int kDBL = kDTR + 2 * kDS; // 80

typedef __attribute__((ext_vector_type(16))) _Float16 v16h;
typedef __attribute__((ext_vector_type(8)))  _Float16 v8h;
typedef __attribute__((ext_vector_type(4)))  _Float16 v4h;
typedef __attribute__((ext_vector_type(8)))  float    v8f;

__device__ __forceinline__ float dsigmoid(float x) { return 1.0f / (1.0f + __expf(-x)); }

// =====================================================================
// FAST WMMA GEMM (no guards): requires M%64==0, N%64==0, K%64==0.
// C[M,N] = A[M,K] @ W[N,K]^T (+bias) (+Res).  f32 in/out, f16 WMMA.
// 256 threads = 8 waves; tile 64x64; K-stage 64 (2 WMMA k-substeps).
// Register-staged double buffering: next stage's global loads are issued
// before the WMMAs of the current stage, overlapping HBM with matrix pipe.
// =====================================================================
constexpr int kStrF = 72;   // halves per LDS row for K=64 stage (144B, 16B-aligned)

template <bool RES>
__global__ void __launch_bounds__(256)
gemm_wmma_fast(const float* __restrict__ A, int lda,
               const float* __restrict__ W,
               const float* __restrict__ bias,
               const float* Res, int ldr,
               float* __restrict__ C, int ldc,
               int M, int N, int K)
{
    __shared__ __attribute__((aligned(16))) _Float16 As[64 * kStrF];
    __shared__ __attribute__((aligned(16))) _Float16 Bs[64 * kStrF];

    const int tid  = threadIdx.x;
    const int lane = tid & 31;
    const int wave = tid >> 5;
    const int mt   = wave & 3;    // 4 M sub-tiles
    const int nt   = wave >> 2;   // 2 N sub-tile pairs
    const int tileM = blockIdx.y * 64;
    const int tileN = blockIdx.x * 64;

    v8f acc0 = {};
    v8f acc1 = {};

    // per-thread staging coordinates: 64 rows x 16 float4 per matrix
    int srow[4], scol[4];
    #pragma unroll
    for (int u = 0; u < 4; ++u) {
        const int unit = tid + u * 256;  // 0..1023
        srow[u] = unit >> 4;             // 0..63
        scol[u] = (unit & 15) << 2;      // 0,4,...,60
    }

    float4 ra[4], rw[4];
    auto load_stage = [&](int kk) {
        #pragma unroll
        for (int u = 0; u < 4; ++u) {
            ra[u] = *(const float4*)(A + (size_t)(tileM + srow[u]) * lda + kk + scol[u]);
            rw[u] = *(const float4*)(W + (size_t)(tileN + srow[u]) * K   + kk + scol[u]);
        }
    };
    auto store_stage = [&]() {
        #pragma unroll
        for (int u = 0; u < 4; ++u) {
            v4h ha = {(_Float16)ra[u].x, (_Float16)ra[u].y, (_Float16)ra[u].z, (_Float16)ra[u].w};
            *(v4h*)&As[srow[u] * kStrF + scol[u]] = ha;
            v4h hw = {(_Float16)rw[u].x, (_Float16)rw[u].y, (_Float16)rw[u].z, (_Float16)rw[u].w};
            *(v4h*)&Bs[srow[u] * kStrF + scol[u]] = hw;
        }
    };

    // fragment source offsets (contiguous k order for A and B: lane half h holds k 16h..16h+15)
    const int arow  = mt * 16 + (lane & 15);
    const int acOff = (lane & 16) ? 16 : 0;   // A: K chunk offset (halves)
    const int bcol  = nt * 32 + (lane & 15);
    const int bkOff = (lane & 16) ? 16 : 0;   // B: K chunk offset (halves)

    load_stage(0);
    for (int kk = 0; kk < K; kk += 64) {
        __syncthreads();
        store_stage();
        __syncthreads();
        if (kk + 64 < K) load_stage(kk + 64);   // in flight during WMMAs

        #pragma unroll
        for (int ks = 0; ks < 64; ks += 32) {
            // A 16x32: lane<16 -> K 0..15; lane>=16 -> K 16..31
            v8h alo = *(const v8h*)&As[arow * kStrF + ks + acOff];
            v8h ahi = *(const v8h*)&As[arow * kStrF + ks + acOff + 8];
            v16h af = __builtin_shufflevector(alo, ahi, 0,1,2,3,4,5,6,7,8,9,10,11,12,13,14,15);
            // B 32x16: lane<16 -> K 0..15; lane>=16 -> K 16..31
            v8h b0l = *(const v8h*)&Bs[bcol * kStrF + ks + bkOff];
            v8h b0h = *(const v8h*)&Bs[bcol * kStrF + ks + bkOff + 8];
            v16h bf0 = __builtin_shufflevector(b0l, b0h, 0,1,2,3,4,5,6,7,8,9,10,11,12,13,14,15);
            v8h b1l = *(const v8h*)&Bs[(bcol + 16) * kStrF + ks + bkOff];
            v8h b1h = *(const v8h*)&Bs[(bcol + 16) * kStrF + ks + bkOff + 8];
            v16h bf1 = __builtin_shufflevector(b1l, b1h, 0,1,2,3,4,5,6,7,8,9,10,11,12,13,14,15);

            acc0 = __builtin_amdgcn_wmma_f32_16x16x32_f16(false, af, false, bf0, (short)0, acc0, false, false);
            acc1 = __builtin_amdgcn_wmma_f32_16x16x32_f16(false, af, false, bf1, (short)0, acc1, false, false);
        }
    }

    // C/D 16x16 layout: VGPR r -> M = r + (lane>=16 ? 8:0), N = lane&15
    const int row0 = tileM + mt * 16 + ((lane & 16) ? 8 : 0);
    const int col0 = tileN + nt * 32 + (lane & 15);
    #pragma unroll
    for (int r = 0; r < 8; ++r) {
        const int row = row0 + r;
        #pragma unroll
        for (int t = 0; t < 2; ++t) {
            const int col = col0 + t * 16;
            float v = t ? acc1[r] : acc0[r];
            if (bias) v += bias[col];
            if (RES) v += Res[(size_t)row * ldr + col];
            C[(size_t)row * ldc + col] = v;
        }
    }
}

// =====================================================================
// EDGE WMMA GEMM (guarded): used only for the two small odd-shape GEMMs
// (x-proj N=80, dt-proj K=48).  K-step 32.  ACT: 0=none, 2=softplus.
// =====================================================================
constexpr int kLdsStr = 40;  // halves per LDS row (80B, 16B-aligned)

template <int ACT>
__global__ void __launch_bounds__(256)
gemm_wmma_edge(const float* __restrict__ A, int lda,
               const float* __restrict__ W,
               const float* __restrict__ bias,
               float* __restrict__ C, int ldc,
               int M, int N, int K)
{
    __shared__ __attribute__((aligned(16))) _Float16 As[64 * kLdsStr];
    __shared__ __attribute__((aligned(16))) _Float16 Bs[64 * kLdsStr];

    const int tid  = threadIdx.x;
    const int lane = tid & 31;
    const int wave = tid >> 5;
    const int mt   = wave & 3;
    const int nt   = wave >> 2;
    const int tileM = blockIdx.y * 64;
    const int tileN = blockIdx.x * 64;

    v8f acc0 = {};
    v8f acc1 = {};

    const int arow  = mt * 16 + (lane & 15);
    const int acOff = (lane & 16) ? 16 : 0;
    const int bcol  = nt * 32 + (lane & 15);
    const int bkOff = (lane & 16) ? 16 : 0;

    for (int kk = 0; kk < K; kk += 32) {
        __syncthreads();
        #pragma unroll
        for (int u = 0; u < 2; ++u) {
            const int unit = tid + u * 256;
            const int row  = unit >> 3;
            const int c0   = (unit & 7) * 4;
            {
                const int gr = tileM + row;
                const bool rin = gr < M;
                const float* src = A + (size_t)gr * lda + kk + c0;
                float4 v;
                if (rin && (kk + c0 + 4 <= K)) v = *(const float4*)src;
                else {
                    v.x = (rin && kk + c0 + 0 < K) ? src[0] : 0.f;
                    v.y = (rin && kk + c0 + 1 < K) ? src[1] : 0.f;
                    v.z = (rin && kk + c0 + 2 < K) ? src[2] : 0.f;
                    v.w = (rin && kk + c0 + 3 < K) ? src[3] : 0.f;
                }
                v4h h = {(_Float16)v.x, (_Float16)v.y, (_Float16)v.z, (_Float16)v.w};
                *(v4h*)&As[row * kLdsStr + c0] = h;
            }
            {
                const int gn = tileN + row;
                const bool nin = gn < N;
                const float* src = W + (size_t)gn * K + kk + c0;
                float4 v;
                if (nin && (kk + c0 + 4 <= K)) v = *(const float4*)src;
                else {
                    v.x = (nin && kk + c0 + 0 < K) ? src[0] : 0.f;
                    v.y = (nin && kk + c0 + 1 < K) ? src[1] : 0.f;
                    v.z = (nin && kk + c0 + 2 < K) ? src[2] : 0.f;
                    v.w = (nin && kk + c0 + 3 < K) ? src[3] : 0.f;
                }
                v4h h = {(_Float16)v.x, (_Float16)v.y, (_Float16)v.z, (_Float16)v.w};
                *(v4h*)&Bs[row * kLdsStr + c0] = h;
            }
        }
        __syncthreads();

        v8h alo = *(const v8h*)&As[arow * kLdsStr + acOff];
        v8h ahi = *(const v8h*)&As[arow * kLdsStr + acOff + 8];
        v16h af = __builtin_shufflevector(alo, ahi, 0,1,2,3,4,5,6,7,8,9,10,11,12,13,14,15);
        v8h b0l = *(const v8h*)&Bs[bcol * kLdsStr + bkOff];
        v8h b0h = *(const v8h*)&Bs[bcol * kLdsStr + bkOff + 8];
        v16h bf0 = __builtin_shufflevector(b0l, b0h, 0,1,2,3,4,5,6,7,8,9,10,11,12,13,14,15);
        v8h b1l = *(const v8h*)&Bs[(bcol + 16) * kLdsStr + bkOff];
        v8h b1h = *(const v8h*)&Bs[(bcol + 16) * kLdsStr + bkOff + 8];
        v16h bf1 = __builtin_shufflevector(b1l, b1h, 0,1,2,3,4,5,6,7,8,9,10,11,12,13,14,15);

        acc0 = __builtin_amdgcn_wmma_f32_16x16x32_f16(false, af, false, bf0, (short)0, acc0, false, false);
        acc1 = __builtin_amdgcn_wmma_f32_16x16x32_f16(false, af, false, bf1, (short)0, acc1, false, false);
    }

    const int row0 = tileM + mt * 16 + ((lane & 16) ? 8 : 0);
    const int col0 = tileN + nt * 32 + (lane & 15);
    #pragma unroll
    for (int r = 0; r < 8; ++r) {
        const int row = row0 + r;
        if (row >= M) continue;
        #pragma unroll
        for (int t = 0; t < 2; ++t) {
            const int col = col0 + t * 16;
            if (col >= N) continue;
            float v = t ? acc1[r] : acc0[r];
            if (bias) v += bias[col];
            if (ACT == 2) v = (v > 20.f) ? v : __logf(1.f + __expf(v));
            C[(size_t)row * ldc + col] = v;
        }
    }
}

// =====================================================================
// LayerNorm over last dim (kDIM), one workgroup per row
// =====================================================================
__global__ void __launch_bounds__(256)
ln_kernel(const float* __restrict__ in, const float* __restrict__ g,
          const float* __restrict__ bt, float* __restrict__ out)
{
    __shared__ float s1[256], s2[256];
    const int row = blockIdx.x;
    const float* x = in + (size_t)row * kDIM;
    float a = 0.f, b = 0.f;
    for (int i = threadIdx.x; i < kDIM; i += 256) { float v = x[i]; a += v; b += v * v; }
    s1[threadIdx.x] = a; s2[threadIdx.x] = b;
    __syncthreads();
    for (int o = 128; o > 0; o >>= 1) {
        if (threadIdx.x < o) { s1[threadIdx.x] += s1[threadIdx.x + o]; s2[threadIdx.x] += s2[threadIdx.x + o]; }
        __syncthreads();
    }
    const float mean = s1[0] / kDIM;
    const float var  = s2[0] / kDIM - mean * mean;
    const float rs   = rsqrtf(var + 1e-5f);
    for (int i = threadIdx.x; i < kDIM; i += 256)
        out[(size_t)row * kDIM + i] = (x[i] - mean) * rs * g[i] + bt[i];
}

// =====================================================================
// Depthwise causal conv1d (DC=4) + bias + SiLU.  xi = xz[:, :, 0:DI]
// =====================================================================
__global__ void conv_silu_kernel(const float* __restrict__ xz, const float* __restrict__ cw,
                                 const float* __restrict__ cb, float* __restrict__ xc, int L)
{
    const int idx = blockIdx.x * blockDim.x + threadIdx.x;
    const int total = kB * L * kDI;
    if (idx >= total) return;
    const int d = idx % kDI;
    const int l = (idx / kDI) % L;
    const int b = idx / (kDI * L);
    const float* base = xz + (size_t)b * L * (2 * kDI) + d;
    float acc = cb[d];
    #pragma unroll
    for (int j = 0; j < kDC; ++j) {
        const int ll = l - (kDC - 1) + j;
        if (ll >= 0) acc += base[(size_t)ll * (2 * kDI)] * cw[d * kDC + j];
    }
    xc[idx] = acc * dsigmoid(acc);
}

// =====================================================================
// Selective scan: one thread per (b, d); h[16] in registers; B/C rows
// fetched as float4.  yb aliases dtb (owner-thread read-then-write).
// =====================================================================
__global__ void scan_kernel(const float* dtb, const float* __restrict__ xcb,
                            const float* __restrict__ xzb, const float* __restrict__ dblb,
                            const float* __restrict__ A_log, const float* __restrict__ Dp,
                            float* yb, int L)
{
    const int idx = blockIdx.x * blockDim.x + threadIdx.x;
    if (idx >= kB * kDI) return;
    const int d = idx % kDI;
    const int b = idx / kDI;
    float Ar[kDS];
    #pragma unroll
    for (int s = 0; s < kDS; ++s) Ar[s] = -__expf(A_log[d * kDS + s]);
    const float Dv = Dp[d];
    float h[kDS] = {0.f};
    for (int l = 0; l < L; ++l) {
        const size_t row = (size_t)b * L + l;
        const float dtv = dtb[row * kDI + d];
        const float xcv = xcb[row * kDI + d];
        const float zv  = xzb[row * (2 * kDI) + kDI + d];
        const float* bl = dblb + row * kDBL;
        float bm[kDS], cm[kDS];
        #pragma unroll
        for (int j = 0; j < kDS / 4; ++j) {
            float4 t0 = *(const float4*)(bl + kDTR + 4 * j);
            bm[4*j+0] = t0.x; bm[4*j+1] = t0.y; bm[4*j+2] = t0.z; bm[4*j+3] = t0.w;
            float4 t1 = *(const float4*)(bl + kDTR + kDS + 4 * j);
            cm[4*j+0] = t1.x; cm[4*j+1] = t1.y; cm[4*j+2] = t1.z; cm[4*j+3] = t1.w;
        }
        const float dx = dtv * xcv;
        float y = 0.f;
        #pragma unroll
        for (int s = 0; s < kDS; ++s) {
            const float hp = __expf(dtv * Ar[s]) * h[s] + dx * bm[s];
            h[s] = hp;
            y += hp * cm[s];
        }
        y += Dv * xcv;
        y *= zv * dsigmoid(zv);
        yb[row * kDI + d] = y;
    }
}

// =====================================================================
// Block-local attention core: each query attends to 2-3 contiguous keys.
// One thread per (b, q, h); K and V interleaved in KV (row = [k | v]).
// =====================================================================
__global__ void attn_kernel(const float* __restrict__ Q, const float* __restrict__ KV,
                            float* __restrict__ O)
{
    const int idx = blockIdx.x * blockDim.x + threadIdx.x;
    if (idx >= kB * kNQ * kNH) return;
    const int h = idx % kNH;
    const int q = (idx / kNH) % kNQ;
    const int b = idx / (kNH * kNQ);
    const int kpq = kL / kNQ, extra = kL % kNQ;      // 2, 256
    const int start = q * kpq + (q < extra ? q : extra);
    const int cnt   = kpq + (q < extra ? 1 : 0);     // 2 or 3
    const float scale = 0.144337567297406441f;       // 1/sqrt(48)
    const float* qp = Q + (size_t)(b * kNQ + q) * kDIM + h * kHD;

    float s[3];
    float mx = -1e30f;
    for (int j = 0; j < cnt; ++j) {
        const float* kp = KV + (size_t)(b * kL + start + j) * (2 * kDIM) + h * kHD;
        float dacc = 0.f;
        #pragma unroll
        for (int t = 0; t < kHD; ++t) dacc += qp[t] * kp[t];
        s[j] = dacc * scale;
        mx = fmaxf(mx, s[j]);
    }
    float den = 0.f;
    for (int j = 0; j < cnt; ++j) { s[j] = __expf(s[j] - mx); den += s[j]; }
    const float inv = 1.f / den;
    float o[kHD];
    #pragma unroll
    for (int t = 0; t < kHD; ++t) o[t] = 0.f;
    for (int j = 0; j < cnt; ++j) {
        const float* vp = KV + (size_t)(b * kL + start + j) * (2 * kDIM) + kDIM + h * kHD;
        const float a = s[j] * inv;
        #pragma unroll
        for (int t = 0; t < kHD; ++t) o[t] += a * vp[t];
    }
    float* op = O + (size_t)(b * kNQ + q) * kDIM + h * kHD;
    #pragma unroll
    for (int t = 0; t < kHD; ++t) op[t] = o[t];
}

// ---------------- small elementwise kernels ----------------
__global__ void bcast_q_kernel(const float* __restrict__ q, float* __restrict__ X)
{
    const int idx = blockIdx.x * blockDim.x + threadIdx.x;
    const int n = kNQ * kDIM;
    if (idx >= kB * n) return;
    X[idx] = q[idx % n];
}

__global__ void flip_kernel(const float* __restrict__ in, float* __restrict__ out)
{
    const int idx = blockIdx.x * blockDim.x + threadIdx.x;
    if (idx >= kB * kL * kDIM) return;
    const int d = idx % kDIM;
    const int l = (idx / kDIM) % kL;
    const int b = idx / (kDIM * kL);
    out[idx] = in[((size_t)b * kL + (kL - 1 - l)) * kDIM + d];
}

// c = fwd + flip_L(bwd_unflipped) + feats
__global__ void add3flip_kernel(const float* __restrict__ fwd, const float* __restrict__ bwdU,
                                const float* __restrict__ feats, float* __restrict__ c)
{
    const int idx = blockIdx.x * blockDim.x + threadIdx.x;
    if (idx >= kB * kL * kDIM) return;
    const int d = idx % kDIM;
    const int l = (idx / kDIM) % kL;
    const int b = idx / (kDIM * kL);
    c[idx] = fwd[idx] + bwdU[((size_t)b * kL + (kL - 1 - l)) * kDIM + d] + feats[idx];
}

// =====================================================================
// host orchestration
// =====================================================================
extern "C" void kernel_launch(void* const* d_in, const int* in_sizes, int n_in,
                              void* d_out, int out_size, void* d_ws, size_t ws_size,
                              hipStream_t stream)
{
    const float* img_emb    = (const float*)d_in[0];
    const float* queries    = (const float*)d_in[1];
    const float* vproj_w    = (const float*)d_in[2];
    const float* vproj_b    = (const float*)d_in[3];
    const float* lnv_g      = (const float*)d_in[4];
    const float* lnv_b      = (const float*)d_in[5];
    const float* m_in_w     = (const float*)d_in[6];
    const float* m_conv_w   = (const float*)d_in[7];
    const float* m_conv_b   = (const float*)d_in[8];
    const float* m_xproj_w  = (const float*)d_in[9];
    const float* m_dt_w     = (const float*)d_in[10];
    const float* m_dt_b     = (const float*)d_in[11];
    const float* m_A_log    = (const float*)d_in[12];
    const float* m_D        = (const float*)d_in[13];
    const float* m_out_w    = (const float*)d_in[14];
    const float* seq_g      = (const float*)d_in[15];
    const float* seq_b      = (const float*)d_in[16];
    const float* attn_in_w  = (const float*)d_in[17];
    const float* attn_in_b  = (const float*)d_in[18];
    const float* attn_out_w = (const float*)d_in[19];
    const float* attn_out_b = (const float*)d_in[20];
    const float* cross_w    = (const float*)d_in[21];
    const float* cross_b    = (const float*)d_in[22];
    (void)in_sizes; (void)n_in; (void)out_size; (void)ws_size;

    // ---- workspace carve-out (floats) ----
    float* w = (float*)d_ws;
    size_t off = 0;
    auto take = [&](size_t n) { float* p = w + off; off += n; return p; };
    float* feats  = take((size_t)kB * kL * kDIM);
    float* fwdb   = take((size_t)kB * kL * kDIM);
    float* flipin = take((size_t)kB * kL * kDIM);
    float* tmpO   = take((size_t)kB * kL * kDIM);
    float* cbuf   = take((size_t)kB * kL * kDIM);
    float* xz     = take((size_t)kB * kL * 2 * kDI);   // mamba scratch (encoder-sized)
    float* xc     = take((size_t)kB * kL * kDI);
    float* dbl    = take((size_t)kB * kL * kDBL);
    float* dtbuf  = take((size_t)kB * kL * kDI);       // also scan output y (aliased)
    float* X      = take((size_t)kB * kNQ * kDIM);
    float* LNX    = take((size_t)kB * kNQ * kDIM);
    float* XCAT   = take((size_t)kB * kNQ * 2 * kDIM); // [x | score]
    float* Qb     = take((size_t)kB * kNQ * kDIM);
    float* KVb    = take((size_t)kB * kL * 2 * kDIM);  // [k | v]
    float* ATMP   = take((size_t)kB * kNQ * kDIM);

    // fast GEMM (M,N,K all % 64 == 0)
    auto gemm = [&](const float* A, int lda, const float* W, const float* bias,
                    float* C, int ldc, int M, int N, int K) {
        dim3 g(N / 64, M / 64);
        gemm_wmma_fast<false><<<g, 256, 0, stream>>>(A, lda, W, bias, nullptr, 0, C, ldc, M, N, K);
    };
    auto gemm_res = [&](const float* A, int lda, const float* W, const float* bias,
                        const float* Res, int ldr, float* C, int ldc, int M, int N, int K) {
        dim3 g(N / 64, M / 64);
        gemm_wmma_fast<true><<<g, 256, 0, stream>>>(A, lda, W, bias, Res, ldr, C, ldc, M, N, K);
    };

    // One Mamba instance: x_in (M=kB*L rows, lda) -> outp (+ optional residual)
    auto run_mamba = [&](const float* xin, int lda_in, int L, int m,
                         float* outp, int ldo, const float* res, int ldr) {
        const int Mr = kB * L;
        // in-proj: xz = x @ in_w^T   (N = 2*DI = 3072, K = 768)
        gemm(xin, lda_in, m_in_w + (size_t)m * 2 * kDI * kDIM, nullptr, xz, 2 * kDI, Mr, 2 * kDI, kDIM);
        // depthwise causal conv + SiLU on xi half
        const int tot = Mr * kDI;
        conv_silu_kernel<<<(tot + 255) / 256, 256, 0, stream>>>(
            xz, m_conv_w + (size_t)m * kDI * kDC, m_conv_b + (size_t)m * kDI, xc, L);
        // x-proj: dbl = xc @ xproj^T (N = 80 -> edge kernel)
        {
            dim3 g((kDBL + 63) / 64, Mr / 64);
            gemm_wmma_edge<0><<<g, 256, 0, stream>>>(
                xc, kDI, m_xproj_w + (size_t)m * kDBL * kDI, nullptr, dbl, kDBL, Mr, kDBL, kDI);
        }
        // dt = softplus(dbl[:, :48] @ dt_w^T + dt_b)  (K = 48 -> edge kernel)
        {
            dim3 g(kDI / 64, Mr / 64);
            gemm_wmma_edge<2><<<g, 256, 0, stream>>>(
                dbl, kDBL, m_dt_w + (size_t)m * kDI * kDTR, m_dt_b + (size_t)m * kDI,
                dtbuf, kDI, Mr, kDI, kDTR);
        }
        // selective scan (writes y over dtbuf)
        scan_kernel<<<(kB * kDI + 255) / 256, 256, 0, stream>>>(
            dtbuf, xc, xz, dbl, m_A_log + (size_t)m * kDI * kDS, m_D + (size_t)m * kDI, dtbuf, L);
        // out-proj (+residual)   N = 768, K = 1536
        if (res) gemm_res(dtbuf, kDI, m_out_w + (size_t)m * kDIM * kDI, nullptr, res, ldr, outp, ldo, Mr, kDIM, kDI);
        else     gemm(dtbuf, kDI, m_out_w + (size_t)m * kDIM * kDI, nullptr, outp, ldo, Mr, kDIM, kDI);
    };

    const int encRows = kB * kL;     // 2048
    const int qryRows = kB * kNQ;    // 768

    // ---- vision projection + LN ----
    gemm(img_emb, kDIN, vproj_w, vproj_b, xz /*scratch pre-LN*/, kDIM, encRows, kDIM, kDIN);
    ln_kernel<<<encRows, 256, 0, stream>>>(xz, lnv_g, lnv_b, feats);

    // ---- bidirectional encoder mambas ----
    run_mamba(feats, kDIM, kL, 0, fwdb, kDIM, nullptr, 0);
    {
        const int tot = kB * kL * kDIM;
        flip_kernel<<<(tot + 255) / 256, 256, 0, stream>>>(feats, flipin);
    }
    run_mamba(flipin, kDIM, kL, 1, tmpO, kDIM, nullptr, 0);
    {
        const int tot = kB * kL * kDIM;
        add3flip_kernel<<<(tot + 255) / 256, 256, 0, stream>>>(fwdb, tmpO, feats, cbuf);
    }

    // ---- queries ----
    {
        const int tot = kB * kNQ * kDIM;
        bcast_q_kernel<<<(tot + 255) / 256, 256, 0, stream>>>(queries, X);
    }

    // ---- decoder blocks ----
    for (int l = 0; l < kNB; ++l) {
        // x = LN(x); x = mamba(x) + x   (residual is the LN'd x per reference)
        ln_kernel<<<qryRows, 256, 0, stream>>>(X, seq_g + (size_t)l * kDIM, seq_b + (size_t)l * kDIM, LNX);
        run_mamba(LNX, kDIM, kNQ, l + 2, XCAT /*x half*/, 2 * kDIM, LNX, kDIM);

        // q from x (XCAT first half), k|v fused from c
        const float* wq = attn_in_w + (size_t)l * 3 * kDIM * kDIM;
        const float* bq = attn_in_b + (size_t)l * 3 * kDIM;
        gemm(XCAT, 2 * kDIM, wq, bq, Qb, kDIM, qryRows, kDIM, kDIM);
        gemm(cbuf, kDIM, wq + (size_t)kDIM * kDIM, bq + kDIM, KVb, 2 * kDIM, encRows, 2 * kDIM, kDIM);

        // block-local attention core (2-3 keys per query)
        {
            const int tot = kB * kNQ * kNH;
            attn_kernel<<<(tot + 255) / 256, 256, 0, stream>>>(Qb, KVb, ATMP);
        }
        // attention output projection -> score half of XCAT
        gemm(ATMP, kDIM, attn_out_w + (size_t)l * kDIM * kDIM, attn_out_b + (size_t)l * kDIM,
             XCAT + kDIM, 2 * kDIM, qryRows, kDIM, kDIM);

        // x = concat([x, score]) @ cross_w^T + cross_b
        float* xdst = (l == kNB - 1) ? (float*)d_out : X;
        gemm(XCAT, 2 * kDIM, cross_w + (size_t)l * kDIM * 2 * kDIM, cross_b + (size_t)l * kDIM,
             xdst, kDIM, qryRows, kDIM, 2 * kDIM);
    }
}
